// ChemResBlock_31903017074941
// MI455X (gfx1250) — matrix operand order repaired and checked by reading, not compiled, not run on hardware
//
#include <hip/hip_runtime.h>

// ---------------------------------------------------------------------------
// ChemResBlock on MI455X (gfx1250, wave32, WMMA)
//   A=2048 atoms, D=O=64, FL=12
//   Strategy:
//     * one-time convert connectivity -> f16 planes connH[f][a][k]  (96 MB,
//       L2-resident for the 3 re-reads)
//     * fold score/bf contraction into W[f][d][o] = pf[o,f,d]*bf[o,f,0]
//     * bond_term precomputed per layer
//     * fused conv kernel: GEMM1 (16x2048x64, f16 WMMA) with ASYNC memory->LDS
//       staging of h, LDS reshape, GEMM2 (16x64x64, f16 WMMA), cross-filter
//       ds_add_f32 reduction, fused epilogue (bond + residual + relu +
//       f16 transpose for the next conv)
// ---------------------------------------------------------------------------

typedef _Float16 h16;
typedef __attribute__((ext_vector_type(16))) _Float16 v16h;
typedef __attribute__((ext_vector_type(8)))  _Float16 v8h;
typedef __attribute__((ext_vector_type(8)))  float    v8f;

constexpr int AN = 2048;   // atoms
constexpr int DN = 64;     // depth
constexpr int FLN = 12;    // filter length
constexpr int KSL = 256;   // K-slice staged in LDS

static __device__ __forceinline__ v16h cat16(v8h lo, v8h hi) {
  return __builtin_shufflevector(lo, hi, 0,1,2,3,4,5,6,7,8,9,10,11,12,13,14,15);
}

static __device__ __forceinline__ void wait_asynccnt0() {
#if defined(__has_builtin)
#if __has_builtin(__builtin_amdgcn_s_wait_asynccnt)
  __builtin_amdgcn_s_wait_asynccnt(0);
  return;
#else
  asm volatile("s_wait_asynccnt 0x0" ::: "memory");
  return;
#endif
#else
  asm volatile("s_wait_asynccnt 0x0" ::: "memory");
#endif
}

// async DMA: 16 bytes global -> LDS, no VGPR round-trip (ASYNCcnt tracked)
static __device__ __forceinline__ void async_g2l_b128(unsigned lds_off,
                                                      const void* gaddr) {
  asm volatile("global_load_async_to_lds_b128 %0, %1, off"
               :: "v"(lds_off), "v"(gaddr)
               : "memory");
}

// ---- prep: connectivity f32 (A,A,FL) -> connH f16 [f][a][k] ---------------
__global__ void k_prep_conn(const float* __restrict__ conn, h16* __restrict__ connH) {
  int i = blockIdx.x * blockDim.x + threadIdx.x;   // one (a,k) pair
  int a = i >> 11, k = i & (AN - 1);
  const float* p = conn + (size_t)(a * AN + k) * FLN;   // 48B aligned
  float4 v0 = *(const float4*)(p + 0);
  float4 v1 = *(const float4*)(p + 4);
  float4 v2 = *(const float4*)(p + 8);
  float vals[FLN] = {v0.x,v0.y,v0.z,v0.w, v1.x,v1.y,v1.z,v1.w, v2.x,v2.y,v2.z,v2.w};
  size_t base = (size_t)a * AN + k;
#pragma unroll
  for (int f = 0; f < FLN; ++f)
    connH[(size_t)f * AN * AN + base] = (h16)vals[f];
}

// ---- prep: WT[f][o][d] = pf[o,f,d] * bf[o,f,0]   (f16, B-load friendly) ---
__global__ void k_prep_wt(const float* __restrict__ pf, const float* __restrict__ bf,
                          h16* __restrict__ WT) {
  int i = blockIdx.x * blockDim.x + threadIdx.x;
  if (i >= FLN * DN * DN) return;
  int d = i & 63; int rem = i >> 6; int f = rem % FLN; int o = rem / FLN;
  float w = pf[((size_t)o * FLN + f) * DN + d] * bf[((size_t)o * FLN + f) * 3 + 0];
  WT[((size_t)f * DN + o) * DN + d] = (h16)w;
}

// ---- prep: bond[a][o] = sum_f bp[a,f,0]*bf[o,f,1] + bp[a,f,1]*bf[o,f,2] ---
__global__ void k_prep_bond(const float* __restrict__ bp, const float* __restrict__ bf,
                            float* __restrict__ bond) {
  int i = blockIdx.x * blockDim.x + threadIdx.x;   // i = a*64 + o
  int o = i & 63, a = i >> 6;
  float s = 0.f;
#pragma unroll
  for (int f = 0; f < FLN; ++f) {
    s += bp[((size_t)a * FLN + f) * 2 + 0] * bf[((size_t)o * FLN + f) * 3 + 1];
    s += bp[((size_t)a * FLN + f) * 2 + 1] * bf[((size_t)o * FLN + f) * 3 + 2];
  }
  bond[i] = s;
}

// ---- prep: hT[d][a] = (f16) x[a][d] ---------------------------------------
__global__ void k_prep_hT(const float* __restrict__ x, h16* __restrict__ hT) {
  int i = blockIdx.x * blockDim.x + threadIdx.x;   // i = a*64 + d
  int d = i & 63, a = i >> 6;
  hT[(size_t)d * AN + a] = (h16)x[i];
}

// ---- fused conv -----------------------------------------------------------
// block = 384 threads = 12 wave32's, wave w owns filter f = w
// grid.x = 128 a-tiles of 16 rows
__launch_bounds__(384)
__global__ void k_conv(const h16* __restrict__ connH,
                       const h16* __restrict__ hT,
                       const h16* __restrict__ WT,
                       const float* __restrict__ bond,
                       const float* __restrict__ xres,   // residual (may be null)
                       float*      __restrict__ outp,    // f32 out (may be null)
                       h16*        __restrict__ hTout) { // next-h f16 (may be null)
  __shared__ h16  hsl[DN * KSL];        // 32 KB   staged hT slice [d][k]
  __shared__ h16  ncT[FLN * 16 * 72];   // 27 KB   per-wave NC reshape tiles
  __shared__ float red[16 * 68];        // 4.25 KB cross-filter reduction

  const int tid  = threadIdx.x;
  const int wave = tid >> 5;            // == filter index f
  const int lane = tid & 31;
  const int lo   = lane & 15;
  const int hi   = lane >> 4;
  const int a0   = blockIdx.x * 16;

  // LDS byte offset of hsl (flat->LDS translation uses addr[31:0])
  const unsigned hsl_base = (unsigned)(unsigned long long)&hsl[0];

  for (int i = tid; i < 16 * 68; i += 384) red[i] = 0.f;

  v8f acc0 = {}, acc1 = {}, acc2 = {}, acc3 = {};

  // A-fragment row base for this wave's filter plane (A layout used here, B uses the same k order:
  // lane lo holds row a0+lo, K = k0+hi*16..+15 contiguous)
  const h16* aprow = connH + (size_t)wave * AN * AN + (size_t)(a0 + lo) * AN + hi * 16;

  for (int ks = 0; ks < AN; ks += KSL) {
    __syncthreads();   // previous slice fully consumed (all ds data fed WMMAs)
    // async DMA stage of hT[:, ks:ks+KSL] into LDS (b128 granules, no VGPRs)
    for (int idx = tid; idx < (DN * KSL / 8); idx += 384) {
      int row = idx >> 5;              // KSL/8 = 32 vectors per row
      int cv  = idx & 31;
      async_g2l_b128(hsl_base + (unsigned)((row * KSL + cv * 8) * sizeof(h16)),
                     hT + (size_t)row * AN + ks + cv * 8);
    }
    wait_asynccnt0();  // this wave's DMAs landed in LDS
    __syncthreads();   // every wave's DMAs landed
    __builtin_prefetch(aprow + ks + KSL, 0, 0);   // speculative next A slice

    for (int k0 = 0; k0 < KSL; k0 += 32) {
      const h16* ap = aprow + ks + k0;
      v16h af = cat16(*(const v8h*)ap, *(const v8h*)(ap + 8));
      // hoist all 4 B fragments, then 4 back-to-back WMMAs
      const h16* bp0 = &hsl[( 0 + lo) * KSL + k0 + hi * 16];
      const h16* bp1 = &hsl[(16 + lo) * KSL + k0 + hi * 16];
      const h16* bp2 = &hsl[(32 + lo) * KSL + k0 + hi * 16];
      const h16* bp3 = &hsl[(48 + lo) * KSL + k0 + hi * 16];
      v16h b0 = cat16(*(const v8h*)bp0, *(const v8h*)(bp0 + 8));
      v16h b1 = cat16(*(const v8h*)bp1, *(const v8h*)(bp1 + 8));
      v16h b2 = cat16(*(const v8h*)bp2, *(const v8h*)(bp2 + 8));
      v16h b3 = cat16(*(const v8h*)bp3, *(const v8h*)(bp3 + 8));
      acc0 = __builtin_amdgcn_wmma_f32_16x16x32_f16(false, af, false, b0, (short)0, acc0, false, false);
      acc1 = __builtin_amdgcn_wmma_f32_16x16x32_f16(false, af, false, b1, (short)0, acc1, false, false);
      acc2 = __builtin_amdgcn_wmma_f32_16x16x32_f16(false, af, false, b2, (short)0, acc2, false, false);
      acc3 = __builtin_amdgcn_wmma_f32_16x16x32_f16(false, af, false, b3, (short)0, acc3, false, false);
    }
  }

  // C-layout -> A-layout reshape through LDS (f32 accum -> f16 [m][d], ld=72)
  {
    h16* nb = &ncT[wave * 16 * 72];
#pragma unroll
    for (int r = 0; r < 8; ++r) {
      int m = r + 8 * hi;
      nb[m * 72 +  0 + lo] = (h16)acc0[r];
      nb[m * 72 + 16 + lo] = (h16)acc1[r];
      nb[m * 72 + 32 + lo] = (h16)acc2[r];
      nb[m * 72 + 48 + lo] = (h16)acc3[r];
    }
  }
  __syncthreads();

  // GEMM2: OUT_f[16x64] = NC_f[16x64] x W_f[64x64]
  v8f o0 = {}, o1 = {}, o2 = {}, o3 = {};
  {
    const h16* nb = &ncT[wave * 16 * 72];
    const h16* wb = WT + (size_t)wave * DN * DN;
#pragma unroll
    for (int kk = 0; kk < DN; kk += 32) {
      const h16* ap = nb + lo * 72 + kk + hi * 16;
      v16h af = cat16(*(const v8h*)ap, *(const v8h*)(ap + 8));
#pragma unroll
      for (int nt = 0; nt < 4; ++nt) {
        const h16* bp = wb + (size_t)(nt * 16 + lo) * DN + kk + hi * 16;
        v16h bfv = cat16(*(const v8h*)bp, *(const v8h*)(bp + 8));
        v8f& oc = (nt == 0 ? o0 : nt == 1 ? o1 : nt == 2 ? o2 : o3);
        oc = __builtin_amdgcn_wmma_f32_16x16x32_f16(
                 false, af, false, bfv, (short)0, oc, false, false);
      }
    }
  }

  // cross-filter reduction: 12 waves ds_add into shared tile
#pragma unroll
  for (int r = 0; r < 8; ++r) {
    int m = r + 8 * hi;
    atomicAdd(&red[m * 68 +  0 + lo], o0[r]);
    atomicAdd(&red[m * 68 + 16 + lo], o1[r]);
    atomicAdd(&red[m * 68 + 32 + lo], o2[r]);
    atomicAdd(&red[m * 68 + 48 + lo], o3[r]);
  }
  __syncthreads();

  // epilogue: + bond (+ residual), relu, emit f32 and/or next-h f16 transpose
  for (int i = tid; i < 16 * DN; i += 384) {
    int m = i >> 6, d = i & 63;
    int gi = (a0 + m) * DN + d;
    float v = red[m * 68 + d] + bond[gi];
    if (xres) v += xres[gi];
    v = v > 0.f ? v : 0.f;
    if (outp)  outp[gi] = v;
    if (hTout) hTout[(size_t)d * AN + a0 + m] = (h16)v;
  }
}

// ---------------------------------------------------------------------------
extern "C" void kernel_launch(void* const* d_in, const int* in_sizes, int n_in,
                              void* d_out, int out_size, void* d_ws, size_t ws_size,
                              hipStream_t stream) {
  (void)in_sizes; (void)n_in; (void)out_size; (void)ws_size;
  const float* x    = (const float*)d_in[0];
  const float* conn = (const float*)d_in[1];
  const float* bp   = (const float*)d_in[2];
  const float* pf0  = (const float*)d_in[3];
  const float* bf0  = (const float*)d_in[4];
  const float* pf1  = (const float*)d_in[5];
  const float* bf1  = (const float*)d_in[6];
  float* out = (float*)d_out;

  char* ws = (char*)d_ws;
  size_t off = 0;
  auto take = [&](size_t bytes) -> char* {
    char* p = ws + off;
    off = (off + bytes + 255) & ~(size_t)255;
    return p;
  };
  h16*   connH = (h16*)  take((size_t)FLN * AN * AN * sizeof(h16));   // 96 MB
  h16*   hTa   = (h16*)  take((size_t)DN * AN * sizeof(h16));         // 256 KB
  h16*   hTb   = (h16*)  take((size_t)DN * AN * sizeof(h16));         // 256 KB
  h16*   WT0   = (h16*)  take((size_t)FLN * DN * DN * sizeof(h16));   // 96 KB
  h16*   WT1   = (h16*)  take((size_t)FLN * DN * DN * sizeof(h16));   // 96 KB
  float* bond0 = (float*)take((size_t)AN * DN * sizeof(float));       // 512 KB
  float* bond1 = (float*)take((size_t)AN * DN * sizeof(float));       // 512 KB

  // one-time preprocessing
  k_prep_conn<<<(AN * AN) / 256, 256, 0, stream>>>(conn, connH);
  k_prep_wt<<<(FLN * DN * DN + 255) / 256, 256, 0, stream>>>(pf0, bf0, WT0);
  k_prep_wt<<<(FLN * DN * DN + 255) / 256, 256, 0, stream>>>(pf1, bf1, WT1);
  k_prep_bond<<<(AN * DN) / 256, 256, 0, stream>>>(bp, bf0, bond0);
  k_prep_bond<<<(AN * DN) / 256, 256, 0, stream>>>(bp, bf1, bond1);
  k_prep_hT<<<(AN * DN) / 256, 256, 0, stream>>>(x, hTa);

  // layer 0:  t = relu(conv(x));  h = relu(conv(t) + x)
  k_conv<<<AN / 16, 384, 0, stream>>>(connH, hTa, WT0, bond0, nullptr, nullptr, hTb);
  k_conv<<<AN / 16, 384, 0, stream>>>(connH, hTb, WT0, bond0, x, nullptr, hTa);
  // layer 1:  t = relu(conv(h));  out = relu(conv(t) + x)
  k_conv<<<AN / 16, 384, 0, stream>>>(connH, hTa, WT1, bond1, nullptr, nullptr, hTb);
  k_conv<<<AN / 16, 384, 0, stream>>>(connH, hTb, WT1, bond1, x, out, nullptr);
}
